// SelfAttention_87316685127722
// MI455X (gfx1250) — matrix operand order repaired and checked by reading, not compiled, not run on hardware
//
#include <hip/hip_runtime.h>
#include <hip/hip_bf16.h>
#include <stdint.h>

// ---------------------------------------------------------------------------
// Types for CDNA5 WMMA (wave32, v_wmma_f32_16x16x32_bf16)
// ---------------------------------------------------------------------------
typedef __bf16 bf16_t;
typedef __attribute__((ext_vector_type(16))) __bf16 v16bf;
typedef __attribute__((ext_vector_type(8)))  __bf16 v8bf;
typedef __attribute__((ext_vector_type(8)))  float  v8f;
typedef __attribute__((ext_vector_type(4)))  int    v4i;

#define AS1 __attribute__((address_space(1)))
#define AS3 __attribute__((address_space(3)))

// Async global->LDS (CDNA5, ASYNCcnt) with synchronous fallback
#if defined(__has_builtin)
#if __has_builtin(__builtin_amdgcn_global_load_async_to_lds_b128) && \
    __has_builtin(__builtin_amdgcn_s_wait_asynccnt)
#define USE_ASYNC_LDS 1
#endif
#endif
#ifndef USE_ASYNC_LDS
#define USE_ASYNC_LDS 0
#endif

__device__ __forceinline__ bf16_t f2bf(float f) {
    unsigned u = __builtin_bit_cast(unsigned, f);
    unsigned r = (u + 0x7FFFu + ((u >> 16) & 1u)) >> 16;
    return __builtin_bit_cast(bf16_t, (unsigned short)r);
}

// 16B global -> LDS staging (async when available)
__device__ __forceinline__ void stage_cp16(bf16_t* ldst, const bf16_t* gsrc) {
#if USE_ASYNC_LDS
    __builtin_amdgcn_global_load_async_to_lds_b128(
        (AS1 v4i*)gsrc, (AS3 v4i*)ldst, /*offset=*/0, /*cpol=*/0);
#else
    *reinterpret_cast<uint4*>(ldst) = *reinterpret_cast<const uint4*>(gsrc);
#endif
}
// 9 async issues per wave per block: wait for the PREVIOUS block's 9 only.
__device__ __forceinline__ void stage_wait_prev() {
#if USE_ASYNC_LDS
    __builtin_amdgcn_s_wait_asynccnt(9);
#endif
}
__device__ __forceinline__ void stage_wait_all() {
#if USE_ASYNC_LDS
    __builtin_amdgcn_s_wait_asynccnt(0);
#endif
}

// A-fragment (16x32, 16-bit): lane<16 holds K{k0..k0+7, k0+16..k0+23} of row
// M=lane%16; lane>=16 holds K{k0+8..15, k0+24..31}.  Two 16B loads.
__device__ __forceinline__ v16bf load_frag_a(const bf16_t* base, int ld,
                                             int row, int k0, int hi) {
    const bf16_t* p = base + (size_t)row * ld + k0 + hi * 8;
    v8bf lo = *reinterpret_cast<const v8bf*>(p);
    v8bf hb = *reinterpret_cast<const v8bf*>(p + 16);
    return __builtin_shufflevector(lo, hb, 0,1,2,3,4,5,6,7,8,9,10,11,12,13,14,15);
}

// B-fragment (32x16, 16-bit): lane = column n; lane<16 holds K{k0..k0+7, k0+16..k0+23},
// lane>=16 holds K{k0+8..15, k0+24..31}, from [n][k] storage.
__device__ __forceinline__ v16bf load_frag_b(const bf16_t* base, int ld,
                                             int col, int k0, int hi) {
    const bf16_t* p = base + (size_t)col * ld + k0 + hi * 8;
    v8bf lo = *reinterpret_cast<const v8bf*>(p);
    v8bf hb = *reinterpret_cast<const v8bf*>(p + 16);
    return __builtin_shufflevector(lo, hb, 0,1,2,3,4,5,6,7,8,9,10,11,12,13,14,15);
}

__device__ __forceinline__ v8f wmma_bf16(v16bf a, v16bf b, v8f c) {
    return __builtin_amdgcn_wmma_f32_16x16x32_bf16(
        /*neg_a=*/false, a, /*neg_b=*/false, b,
        /*c_mod=*/(short)0, c, /*reuse_a=*/false, /*reuse_b=*/false);
}

// ---------------------------------------------------------------------------
// Problem constants
// ---------------------------------------------------------------------------
#define BB   4
#define NN   4096          // H*W
#define CC   256
#define DD   32            // C/8
#define BN   (BB*NN)       // 16384
#define WROWS 320          // 32(f) + 32(g) + 256(h)

// ---------------------------------------------------------------------------
// Kernel 1: convert x -> bf16 (row-major [BN][256]); pack Wf^T|Wg^T|Wh^T
// as bf16 [320][256] (out-channel major, in-channel contiguous).
// ---------------------------------------------------------------------------
__global__ void prep_kernel(const float* __restrict__ x,
                            const float* __restrict__ kf,
                            const float* __restrict__ kg,
                            const float* __restrict__ kh,
                            bf16_t* __restrict__ xb,
                            bf16_t* __restrict__ wt) {
    int i = blockIdx.x * blockDim.x + threadIdx.x;
    if (i < BN * CC) xb[i] = f2bf(x[i]);
    if (i < WROWS * CC) {
        int j = i >> 8;        // output row 0..319
        int c = i & 255;       // input channel
        float v;
        if (j < 32)       v = kf[c * DD + j];
        else if (j < 64)  v = kg[c * DD + (j - 32)];
        else              v = kh[c * CC + (j - 64)];
        wt[i] = f2bf(v);
    }
}

// ---------------------------------------------------------------------------
// Kernel 2: projections.  [BN,256] x [256,{32|32|256}] via bf16 WMMA.
// One wave per 16x16 output tile; K=256 in 8 wmma steps.
// Writes: kb (=f, key) [BN][32] bf16, qb (=g, query) [BN][32] bf16,
//         vt (=h^T) [B][C][N] bf16 (n contiguous).
// ---------------------------------------------------------------------------
__global__ __launch_bounds__(128)
void proj_kernel(const bf16_t* __restrict__ xb,
                 const bf16_t* __restrict__ wt,
                 bf16_t* __restrict__ qb,
                 bf16_t* __restrict__ kb,
                 bf16_t* __restrict__ vt) {
    const int lane = threadIdx.x & 31;
    const int wave = threadIdx.x >> 5;        // 0..3
    const int hi   = lane >> 4;
    const int ln   = lane & 15;
    const int r0   = blockIdx.x * 16;                 // global row base
    const int colTile = blockIdx.y * 4 + wave;        // 0..19
    const int n0   = colTile * 16;

    v8f acc = {};
#pragma unroll
    for (int kk = 0; kk < CC; kk += 32) {
        v16bf a = load_frag_a(xb, CC, r0 + ln, kk, hi);
        v16bf b = load_frag_b(wt, CC, n0 + ln, kk, hi);
        acc = wmma_bf16(a, b, acc);
    }

    const int col = n0 + ln;
#pragma unroll
    for (int r = 0; r < 8; ++r) {
        int q = r0 + r + 8 * hi;                      // global row 0..BN-1
        bf16_t v = f2bf(acc[r]);
        if (col < 32) {                               // f -> key
            kb[(size_t)q * DD + col] = v;
        } else if (col < 64) {                        // g -> query
            qb[(size_t)q * DD + (col - 32)] = v;
        } else {                                      // h -> V transposed
            int b_ = q >> 12, n = q & (NN - 1);
            vt[((size_t)b_ * CC + (col - 64)) * NN + n] = v;
        }
    }
}

// ---------------------------------------------------------------------------
// Kernel 3: fused flash attention + epilogue (out = gamma*o + x).
// Block = 256 threads = 8 waves; each wave owns 16 queries.
// Double-buffered async global->LDS staging of K/V^T blocks overlapped
// with WMMA compute.  Per 64-key block: S=Q@K^T (4 wmma), online softmax,
// P@V (32 wmma).
// ---------------------------------------------------------------------------
__global__ __launch_bounds__(256)
void attn_kernel(const bf16_t* __restrict__ qb,
                 const bf16_t* __restrict__ kb,
                 const bf16_t* __restrict__ vt,
                 const float* __restrict__ x,
                 const float* __restrict__ gamma,
                 float* __restrict__ out) {
    __shared__ __align__(16) bf16_t Ks[2][64 * DD];     //  8 KB: K blocks [64][32]
    __shared__ __align__(16) bf16_t Vs[2][CC * 64];     // 64 KB: V^T blocks [256][64]
    __shared__ __align__(16) bf16_t Ps[8][16 * 64];     // 16 KB: per-wave P tile

    const int lane = threadIdx.x & 31;
    const int wave = threadIdx.x >> 5;
    const int hi   = lane >> 4;
    const int ln   = lane & 15;
    const int b    = blockIdx.x >> 5;                          // batch
    const int q0   = (blockIdx.x & 31) * 128 + wave * 16;      // query base in batch
    const size_t qg = (size_t)b * NN + q0;                     // global row base

    // Per-thread staging assignment (all threads active, EXEC all ones)
    const int krow = threadIdx.x >> 2, kch = threadIdx.x & 3;  // K: 1 x 16B
    const int vc   = threadIdx.x;                              // V: 8 x 16B
    const bf16_t* kbase = kb + (size_t)b * NN * DD;
    const bf16_t* vbase = vt + (size_t)b * CC * NN;

    auto issue_block = [&](int kbk, int buf) {
        stage_cp16(&Ks[buf][krow * DD + kch * 8],
                   kbase + ((size_t)(kbk * 64 + krow)) * DD + kch * 8);
        const bf16_t* vs = vbase + (size_t)vc * NN + kbk * 64;
        bf16_t* vd = &Vs[buf][vc * 64];
#pragma unroll
        for (int j = 0; j < 8; ++j)
            stage_cp16(vd + j * 8, vs + j * 8);
    };

    // Resident Q fragment (16 rows x 32 K)
    v16bf qfrag = load_frag_a(qb + qg * DD, DD, ln, 0, hi);

    v8f O[16] = {};
    float mrow[8], lrow[8];
#pragma unroll
    for (int r = 0; r < 8; ++r) { mrow[r] = -3.0e38f; lrow[r] = 0.f; }
    const float LOG2E = 1.44269504088896340736f;

    issue_block(0, 0);                 // prologue: prefetch first block

    for (int kbk = 0; kbk < NN / 64; ++kbk) {
        // Prefetch next block into the other buffer (its last readers finished
        // before the trailing barrier of iteration kbk-1), then wait for the
        // CURRENT block's staging only.
        if (kbk + 1 < NN / 64) {
            issue_block(kbk + 1, (kbk + 1) & 1);
            stage_wait_prev();
        } else {
            stage_wait_all();
        }
        __syncthreads();

        const bf16_t* Kcur = Ks[kbk & 1];
        const bf16_t* Vcur = Vs[kbk & 1];

        // S = Q @ K^T : 4 tiles of 16 keys, K=32 in one wmma each
        v8f s[4];
#pragma unroll
        for (int j = 0; j < 4; ++j) {
            v16bf bk = load_frag_b(Kcur, DD, j * 16 + ln, 0, hi);
            v8f z = {};
            s[j] = wmma_bf16(qfrag, bk, z);
        }

        // Online softmax.  Row r lives in VGPR r, actual row = r + 8*hi.
        float mnew[8], alpha[8];
#pragma unroll
        for (int r = 0; r < 8; ++r) {
            float mx = s[0][r];
#pragma unroll
            for (int j = 1; j < 4; ++j) mx = fmaxf(mx, s[j][r]);
#pragma unroll
            for (int d = 1; d < 16; d <<= 1) mx = fmaxf(mx, __shfl_xor(mx, d, 32));
            mnew[r]  = fmaxf(mrow[r], mx);
            alpha[r] = exp2f((mrow[r] - mnew[r]) * LOG2E);
        }
#pragma unroll
        for (int r = 0; r < 8; ++r) {
            float rs = 0.f;
#pragma unroll
            for (int j = 0; j < 4; ++j) {
                float p = exp2f((s[j][r] - mnew[r]) * LOG2E);
                s[j][r] = p;
                rs += p;
            }
#pragma unroll
            for (int d = 1; d < 16; d <<= 1) rs += __shfl_xor(rs, d, 32);
            lrow[r] = lrow[r] * alpha[r] + rs;
            mrow[r] = mnew[r];
        }

        // D-layout -> A-layout via per-wave LDS round trip (bf16 P)
        bf16_t* pw = Ps[wave];
#pragma unroll
        for (int j = 0; j < 4; ++j)
#pragma unroll
            for (int r = 0; r < 8; ++r)
                pw[(r + 8 * hi) * 64 + j * 16 + ln] = f2bf(s[j][r]);

        // Rescale accumulator
#pragma unroll
        for (int ct = 0; ct < 16; ++ct)
#pragma unroll
            for (int r = 0; r < 8; ++r)
                O[ct][r] *= alpha[r];

        // O += P @ V : 16 column tiles x 2 K-chunks
        v16bf pa0 = load_frag_a(pw, 64, ln, 0, hi);
        v16bf pa1 = load_frag_a(pw, 64, ln, 32, hi);
#pragma unroll
        for (int ct = 0; ct < 16; ++ct) {
            v16bf b0 = load_frag_b(Vcur, 64, ct * 16 + ln, 0, hi);
            O[ct] = wmma_bf16(pa0, b0, O[ct]);
            v16bf b1 = load_frag_b(Vcur, 64, ct * 16 + ln, 32, hi);
            O[ct] = wmma_bf16(pa1, b1, O[ct]);
        }
        __syncthreads();   // all waves done reading this buffer
    }

    // Epilogue: out = gamma * (O / l) + x
    float g = gamma[0];
    float inv[8];
#pragma unroll
    for (int r = 0; r < 8; ++r) inv[r] = 1.0f / lrow[r];
#pragma unroll
    for (int ct = 0; ct < 16; ++ct) {
        int col = ct * 16 + ln;
#pragma unroll
        for (int r = 0; r < 8; ++r) {
            size_t q   = qg + r + 8 * hi;
            size_t idx = q * CC + col;
            out[idx] = O[ct][r] * inv[r] * g + x[idx];
        }
    }
}

// ---------------------------------------------------------------------------
// Launch: workspace layout (bytes):
//   xb: [0, 8MB)   Qb: [8MB, 9MB)   Kb: [9MB, 10MB)
//   Vt: [10MB, 18MB)   Wt: [18MB, 18MB+160KB)
// ---------------------------------------------------------------------------
extern "C" void kernel_launch(void* const* d_in, const int* in_sizes, int n_in,
                              void* d_out, int out_size, void* d_ws, size_t ws_size,
                              hipStream_t stream) {
    const float* x     = (const float*)d_in[0];
    const float* kf    = (const float*)d_in[1];
    const float* kg    = (const float*)d_in[2];
    const float* kh    = (const float*)d_in[3];
    const float* gamma = (const float*)d_in[4];
    float* out = (float*)d_out;

    char* ws = (char*)d_ws;
    bf16_t* xb = (bf16_t*)(ws);
    bf16_t* qb = (bf16_t*)(ws + ((size_t)8  << 20));
    bf16_t* kb = (bf16_t*)(ws + ((size_t)9  << 20));
    bf16_t* vt = (bf16_t*)(ws + ((size_t)10 << 20));
    bf16_t* wt = (bf16_t*)(ws + ((size_t)18 << 20));

    prep_kernel<<<(BN * CC) / 256, 256, 0, stream>>>(x, kf, kg, kh, xb, wt);
    proj_kernel<<<dim3(BN / 16, 5), 128, 0, stream>>>(xb, wt, qb, kb, vt);
    attn_kernel<<<(BB * NN) / 128, 256, 0, stream>>>(qb, kb, vt, x, gamma, out);
}
